// TAGATLayer_22239340659362
// MI455X (gfx1250) — matrix operand order repaired and checked by reading, not compiled, not run on hardware
//
#include <hip/hip_runtime.h>
#include <hip/hip_bf16.h>

// Problem constants (from reference)
#define N_TOKENS 100000
#define N_ARGS   20000
#define N_EDGES  320000
#define IN_DIM   768
#define HEADS    8
#define OUT_DIM  64
#define ZCOLS    (HEADS * OUT_DIM)   // 512

// GEMM tiling
#define BM 128
#define BN 128
#define BK 32
#define NK (IN_DIM / BK)   // 24 K-iterations
#define LDT 40             // padded LDS row stride in halfwords (80 B): conflict-free b128 reads

typedef __attribute__((ext_vector_type(16))) _Float16 v16h;
typedef __attribute__((ext_vector_type(2)))  __fp16   v2fp16;
typedef __attribute__((ext_vector_type(8)))  float    v8f;

union FragH { v16h v; uint4 q[2]; };
union AccU  { v8f  v; float f[8]; };

__device__ __forceinline__ unsigned pack2h(float a, float b) {
    v2fp16 p = __builtin_amdgcn_cvt_pkrtz(a, b);   // single v_cvt_pkrtz_f16_f32
    union { v2fp16 h; unsigned u; } t; t.h = p;
    return t.u;
}

__device__ __forceinline__ float leaky(float x) { return x > 0.f ? x : 0.01f * x; }

// Order-preserving float<->uint encoding for atomic max on signed floats
__device__ __forceinline__ unsigned f2ord(float f) {
    unsigned u = __float_as_uint(f);
    return (u & 0x80000000u) ? ~u : (u | 0x80000000u);
}
__device__ __forceinline__ float ord2f(unsigned u) {
    return (u & 0x80000000u) ? __uint_as_float(u & 0x7fffffffu) : __uint_as_float(~u);
}

// ---------------------------------------------------------------------------
// Init: zero out + denom, set segment-max to encoded(-inf). Deterministic.
// ---------------------------------------------------------------------------
__global__ void init_kernel(float* __restrict__ out, float* __restrict__ denom,
                            unsigned* __restrict__ m_ord, int nOut, int nSeg) {
    int i = blockIdx.x * blockDim.x + threadIdx.x;
    if (i < nOut) out[i] = 0.f;
    if (i < nSeg) { denom[i] = 0.f; m_ord[i] = 0x007FFFFFu; /* f2ord(-inf) */ }
}

// ---------------------------------------------------------------------------
// z = h @ W^T via v_wmma_f32_16x16x32_f16, f32 accumulate.
// Block tile 128x128, K-step 32. 8 waves, each computes 32(M) x 64(N).
// Register-staged double buffering: global loads for tile k+1 issue before the
// WMMAs on tile k; converts/ds_stores land after, one barrier per iteration.
// blockIdx.x = N block (4 of them) so sibling blocks share h tiles via L2.
// ---------------------------------------------------------------------------
__global__ __launch_bounds__(256) void gemm_z_kernel(const float* __restrict__ h,
                                                     const float* __restrict__ W,
                                                     float* __restrict__ z) {
    __shared__ __align__(16) unsigned short As[2][BM * LDT];
    __shared__ __align__(16) unsigned short Bs[2][BN * LDT];

    const int tid  = threadIdx.x;
    const int lane = tid & 31;
    const int wave = tid >> 5;
    const int wm   = wave & 3;   // 0..3 -> M offset 32*wm
    const int wn   = wave >> 2;  // 0..1 -> N offset 64*wn

    const int colBase = blockIdx.x * BN;   // output cols (W rows); gridDim.x == 4
    const int rowBase = blockIdx.y * BM;   // token rows

    const v8f vzero = {0.f,0.f,0.f,0.f,0.f,0.f,0.f,0.f};
    v8f acc[2][4];
#pragma unroll
    for (int i = 0; i < 2; i++)
#pragma unroll
        for (int j = 0; j < 4; j++) acc[i][j] = vzero;

    // loader: 8 k-threads (float4 each) x 32 rows per pass
    const int kt = tid & 7;
    const int rt = tid >> 3;
    const int ldsw = rt * LDT + kt * 4;    // writer offset within a buffer (halfwords)

    const int lr   = lane & 15;
    const int half = lane >> 4;            // 0 or 1
    const int aOff = half * 8;             // A: K {0..7,16..23} vs {8..15,24..31}
    const int bOff = half * 8;             // B: K {0..7,16..23} vs {8..15,24..31}

    // Clamped A-row bases (avoid per-lane exec predication; select-zero later)
    float azero[4];
    const float* aptr[4]; const float* bptr[4];
#pragma unroll
    for (int p = 0; p < 4; p++) {
        int row  = rowBase + p * 32 + rt;
        azero[p] = (row < N_TOKENS) ? 1.f : 0.f;
        int rowc = (row < N_TOKENS) ? row : (N_TOKENS - 1);
        aptr[p]  = h + (size_t)rowc * IN_DIM + kt * 4;
        bptr[p]  = W + (size_t)(colBase + p * 32 + rt) * IN_DIM + kt * 4;
    }

    // ---- prologue: stage k=0 into buffer 0 ----
    {
        float4 ra[4], rb[4];
#pragma unroll
        for (int p = 0; p < 4; p++) {
            ra[p] = *reinterpret_cast<const float4*>(aptr[p]);
            rb[p] = *reinterpret_cast<const float4*>(bptr[p]);
        }
#pragma unroll
        for (int p = 0; p < 4; p++) {
            float s = azero[p];
            uint2 wa = make_uint2(pack2h(ra[p].x * s, ra[p].y * s),
                                  pack2h(ra[p].z * s, ra[p].w * s));
            uint2 wb = make_uint2(pack2h(rb[p].x, rb[p].y),
                                  pack2h(rb[p].z, rb[p].w));
            *reinterpret_cast<uint2*>(&As[0][p * 32 * LDT + ldsw]) = wa;
            *reinterpret_cast<uint2*>(&Bs[0][p * 32 * LDT + ldsw]) = wb;
        }
    }
    __syncthreads();

    for (int it = 0; it < NK; ++it) {
        const int cur = it & 1;
        const int nxt = cur ^ 1;
        const int kn  = (it + 1) * BK;
        const bool more = (kn < IN_DIM);

        // ---- 1. issue next-tile global loads (no waits yet) ----
        float4 ra[4], rb[4];
        if (more) {
#pragma unroll
            for (int p = 0; p < 4; p++) {
                ra[p] = *reinterpret_cast<const float4*>(aptr[p] + kn);
                rb[p] = *reinterpret_cast<const float4*>(bptr[p] + kn);
            }
        }

        // ---- 2. fragments from current buffer + 8 WMMAs ----
        v16h a[2], b[4];
#pragma unroll
        for (int i = 0; i < 2; i++) {
            const unsigned short* p = &As[cur][(wm * 32 + i * 16 + lr) * LDT + aOff];
            FragH f;
            f.q[0] = *reinterpret_cast<const uint4*>(p);        // K lo 8
            f.q[1] = *reinterpret_cast<const uint4*>(p + 16);   // K hi 8
            a[i] = f.v;
        }
#pragma unroll
        for (int j = 0; j < 4; j++) {
            const unsigned short* p = &Bs[cur][(wn * 64 + j * 16 + lr) * LDT + bOff];
            FragH f;
            f.q[0] = *reinterpret_cast<const uint4*>(p);        // K lo 8
            f.q[1] = *reinterpret_cast<const uint4*>(p + 16);   // K hi 8
            b[j] = f.v;
        }
#pragma unroll
        for (int i = 0; i < 2; i++)
#pragma unroll
            for (int j = 0; j < 4; j++)
                acc[i][j] = __builtin_amdgcn_wmma_f32_16x16x32_f16(
                    false, a[i], false, b[j], (short)0, acc[i][j], false, false);

        // ---- 3. convert + store staged tile into the other buffer ----
        if (more) {
#pragma unroll
            for (int p = 0; p < 4; p++) {
                float s = azero[p];
                uint2 wa = make_uint2(pack2h(ra[p].x * s, ra[p].y * s),
                                      pack2h(ra[p].z * s, ra[p].w * s));
                uint2 wb = make_uint2(pack2h(rb[p].x, rb[p].y),
                                      pack2h(rb[p].z, rb[p].w));
                *reinterpret_cast<uint2*>(&As[nxt][p * 32 * LDT + ldsw]) = wa;
                *reinterpret_cast<uint2*>(&Bs[nxt][p * 32 * LDT + ldsw]) = wb;
            }
        }
        __syncthreads();
    }

    // ---- store: C/D layout — VGPR r: M = r + 8*(lane>=16), N = lane%16 ----
    const int lcol = lane & 15;
    const int rsel = half * 8;
#pragma unroll
    for (int i = 0; i < 2; i++) {
#pragma unroll
        for (int j = 0; j < 4; j++) {
            AccU u; u.v = acc[i][j];
            int gcol = colBase + wn * 64 + j * 16 + lcol;
#pragma unroll
            for (int r = 0; r < 8; r++) {
                int grow = rowBase + wm * 32 + i * 16 + r + rsel;
                if (grow < N_TOKENS)
                    z[(size_t)grow * ZCOLS + gcol] = u.f[r];
            }
        }
    }
}

// ---------------------------------------------------------------------------
// s_node[n,h] = sum_d z[n,h,d] * a_src[d]. One wave per token (4 lanes/head).
// ---------------------------------------------------------------------------
__global__ __launch_bounds__(256) void snode_kernel(const float* __restrict__ z,
                                                    const float* __restrict__ a_attn,
                                                    float* __restrict__ s_node) {
    int token = (blockIdx.x * blockDim.x + threadIdx.x) >> 5;
    int lane  = threadIdx.x & 31;
    if (token >= N_TOKENS) return;   // uniform per wave (grid is exact anyway)

    const float* zr = z + (size_t)token * ZCOLS + lane * 16;
    const int dbase = (lane * 16) & (OUT_DIM - 1);
    float p = 0.f;
#pragma unroll
    for (int i = 0; i < 16; i += 4) {
        float4 v = *reinterpret_cast<const float4*>(zr + i);
        p += v.x * a_attn[dbase + i + 0] + v.y * a_attn[dbase + i + 1]
           + v.z * a_attn[dbase + i + 2] + v.w * a_attn[dbase + i + 3];
    }
    p += __shfl_xor(p, 1, 32);
    p += __shfl_xor(p, 2, 32);
    if ((lane & 3) == 0)
        s_node[(size_t)token * HEADS + (lane >> 2)] = p;
}

// ---------------------------------------------------------------------------
// Segment max of leaky(s_node[src]) per (dst, head) via ordered-uint atomicMax
// ---------------------------------------------------------------------------
__global__ void seg_max_kernel(const float* __restrict__ s_node,
                               const int* __restrict__ esrc, const int* __restrict__ edst,
                               unsigned* __restrict__ m_ord) {
    int gid = blockIdx.x * blockDim.x + threadIdx.x;
    if (gid >= N_EDGES * HEADS) return;
    int e = gid >> 3, hd = gid & 7;
    float v = leaky(s_node[(size_t)esrc[e] * HEADS + hd]);
    atomicMax(&m_ord[(size_t)edst[e] * HEADS + hd], f2ord(v));
}

// ---------------------------------------------------------------------------
// denom[dst,h] = sum exp(e - m)
// ---------------------------------------------------------------------------
__global__ void seg_sum_kernel(const float* __restrict__ s_node,
                               const int* __restrict__ esrc, const int* __restrict__ edst,
                               const unsigned* __restrict__ m_ord,
                               float* __restrict__ denom) {
    int gid = blockIdx.x * blockDim.x + threadIdx.x;
    if (gid >= N_EDGES * HEADS) return;
    int e = gid >> 3, hd = gid & 7;
    size_t idx = (size_t)edst[e] * HEADS + hd;
    float m  = ord2f(m_ord[idx]);
    float ex = __expf(leaky(s_node[(size_t)esrc[e] * HEADS + hd]) - m);
    atomicAdd(&denom[idx], ex);
}

// ---------------------------------------------------------------------------
// out[dst,h,:] += alpha * z[src,h,:]. One wave per edge; lane covers 16 elems.
// ---------------------------------------------------------------------------
__global__ __launch_bounds__(256) void scatter_kernel(const float* __restrict__ z,
                                                      const float* __restrict__ s_node,
                                                      const int* __restrict__ esrc,
                                                      const int* __restrict__ edst,
                                                      const unsigned* __restrict__ m_ord,
                                                      const float* __restrict__ denom,
                                                      float* __restrict__ out) {
    int gid = blockIdx.x * blockDim.x + threadIdx.x;
    if (gid >= N_EDGES * 32) return;
    int e   = gid >> 5;
    int sub = gid & 31;         // head = sub/4, 16-element chunk = sub%4
    int hd  = sub >> 2;

    int src = esrc[e], dst = edst[e];
    size_t idx = (size_t)dst * HEADS + hd;
    float m = ord2f(m_ord[idx]);
    float d = denom[idx];
    d = (d == 0.f) ? 1.f : d;
    float alpha = __expf(leaky(s_node[(size_t)src * HEADS + hd]) - m) / d;

    const float* zp = z   + (size_t)src * ZCOLS + sub * 16;
    float*       op = out + (size_t)dst * ZCOLS + sub * 16;
#pragma unroll
    for (int i = 0; i < 16; i += 4) {
        float4 v = *reinterpret_cast<const float4*>(zp + i);
        atomicAdd(op + i + 0, alpha * v.x);
        atomicAdd(op + i + 1, alpha * v.y);
        atomicAdd(op + i + 2, alpha * v.z);
        atomicAdd(op + i + 3, alpha * v.w);
    }
}

// ---------------------------------------------------------------------------
// Launch
// ---------------------------------------------------------------------------
extern "C" void kernel_launch(void* const* d_in, const int* in_sizes, int n_in,
                              void* d_out, int out_size, void* d_ws, size_t ws_size,
                              hipStream_t stream) {
    const float* h      = (const float*)d_in[0];
    const float* W_fc   = (const float*)d_in[1];
    const float* a_attn = (const float*)d_in[2];
    const int*   esrc   = (const int*)d_in[3];
    const int*   edst   = (const int*)d_in[4];
    float* out = (float*)d_out;

    // Workspace layout (needs ~209.3 MB)
    char* ws = (char*)d_ws;
    float*    z      = (float*)ws;                                     // 100000*512*4 = 204.8 MB
    float*    s_node = (float*)(ws + (size_t)N_TOKENS * ZCOLS * 4);    // 3.2 MB
    unsigned* m_ord  = (unsigned*)(ws + (size_t)N_TOKENS * ZCOLS * 4
                                      + (size_t)N_TOKENS * HEADS * 4); // 640 KB
    float*    denom  = (float*)((char*)m_ord + (size_t)N_ARGS * HEADS * 4);

    const int nOut = N_ARGS * ZCOLS;       // 10,240,000
    const int nSeg = N_ARGS * HEADS;       // 160,000

    init_kernel<<<(nOut + 255) / 256, 256, 0, stream>>>(out, denom, m_ord, nOut, nSeg);

    // x = N block (4) so the 4 blocks sharing one h tile are launch-adjacent (L2 reuse)
    dim3 ggrid(ZCOLS / BN, (N_TOKENS + BM - 1) / BM);   // (4, 782)
    gemm_z_kernel<<<ggrid, 256, 0, stream>>>(h, W_fc, z);

    snode_kernel<<<(N_TOKENS * 32 + 255) / 256, 256, 0, stream>>>(z, a_attn, s_node);

    seg_max_kernel<<<(N_EDGES * HEADS + 255) / 256, 256, 0, stream>>>(s_node, esrc, edst, m_ord);
    seg_sum_kernel<<<(N_EDGES * HEADS + 255) / 256, 256, 0, stream>>>(s_node, esrc, edst, m_ord, denom);

    scatter_kernel<<<(N_EDGES * 32 + 255) / 256, 256, 0, stream>>>(z, s_node, esrc, edst,
                                                                   m_ord, denom, out);
}
